// BloomGuidedClassifier_59691455479928
// MI455X (gfx1250) — matrix operand order repaired and checked by reading, not compiled, not run on hardware
//
#include <hip/hip_runtime.h>
#include <hip/hip_bf16.h>
#include <math.h>

typedef __bf16 bf16_t;
typedef __attribute__((ext_vector_type(8)))  __bf16 v8bf;
typedef __attribute__((ext_vector_type(16))) __bf16 v16bf;
typedef __attribute__((ext_vector_type(8)))  float  v8f;

#define BB 8
#define SS 2048
#define TT (BB*SS)
#define DM 768
#define DB 64
#define NH 4
#define HD 192

__device__ __forceinline__ float gelu_f(float x) {
    return 0.5f * x * (1.0f + erff(x * 0.70710678118654752f));
}
__device__ __forceinline__ v8f v8f_zero() {
    v8f v;
#pragma unroll
    for (int i = 0; i < 8; ++i) v[i] = 0.0f;
    return v;
}
// A fragment (16x32 bf16): lane holds row; K chunk {k0+16*hi..+15} contiguous
__device__ __forceinline__ v16bf ld_afrag(const bf16_t* __restrict__ row, int k0, int hi) {
    v8bf x0 = *(const v8bf*)(row + k0 + 16 * hi);
    v8bf x1 = *(const v8bf*)(row + k0 + 16 * hi + 8);
    v16bf f;
#pragma unroll
    for (int i = 0; i < 8; ++i) { f[i] = x0[i]; f[i + 8] = x1[i]; }
    return f;
}
// B fragment (32x16 bf16): p points at 16 contiguous K elements for this lane's column
__device__ __forceinline__ v16bf ld_bfrag(const bf16_t* __restrict__ p) {
    v8bf x0 = *(const v8bf*)(p);
    v8bf x1 = *(const v8bf*)(p + 8);
    v16bf f;
#pragma unroll
    for (int i = 0; i < 8; ++i) { f[i] = x0[i]; f[i + 8] = x1[i]; }
    return f;
}

// ---------------- conversion kernels ----------------
__global__ void k_cvt_bf16(const float* __restrict__ s, bf16_t* __restrict__ d, int n) {
    int i = blockIdx.x * blockDim.x + threadIdx.x;
    if (i < n) d[i] = (bf16_t)s[i];
}
__global__ void k_copy_cvt(const float* __restrict__ s, float* __restrict__ df,
                           bf16_t* __restrict__ db, int n) {
    int i = blockIdx.x * blockDim.x + threadIdx.x;
    if (i < n) { float v = s[i]; df[i] = v; db[i] = (bf16_t)v; }
}

// ---------------- PositionalBloomEncoder ----------------
__global__ void k_encoder(const float* __restrict__ sig, const float* __restrict__ scale_w,
    const float* __restrict__ c0w, const float* __restrict__ c0b,
    const float* __restrict__ c1w, const float* __restrict__ c1b,
    const float* __restrict__ c2w, const float* __restrict__ c2b,
    const float* __restrict__ lng, const float* __restrict__ lnb,
    float* __restrict__ beF, bf16_t* __restrict__ beB) {
    int t = blockIdx.x * blockDim.x + threadIdx.x;
    if (t >= TT) return;
    int b = t / SS, s = t % SS;
    float w0 = scale_w[0], w1 = scale_w[1], w2 = scale_w[2];
    float mx = fmaxf(w0, fmaxf(w1, w2));
    float e0 = expf(w0 - mx), e1 = expf(w1 - mx), e2 = expf(w2 - mx);
    float inv = 1.0f / (e0 + e1 + e2);
    float wsc[3] = {e0 * inv, e1 * inv, e2 * inv};
    float win[3][7];
#pragma unroll
    for (int dk = 0; dk < 7; ++dk) {
        int sp = s + dk - 3;
        bool ok = (sp >= 0 && sp < SS);
#pragma unroll
        for (int ic = 0; ic < 3; ++ic)
            win[ic][dk] = ok ? sig[((size_t)b * SS + sp) * 3 + ic] * wsc[ic] : 0.0f;
    }
    float ch[64];
    for (int oc = 0; oc < 22; ++oc) {            // k=3, pad 1 -> window 2..4
        float a = c0b[oc];
        for (int ic = 0; ic < 3; ++ic)
            for (int dk = 0; dk < 3; ++dk) a += win[ic][dk + 2] * c0w[(oc * 3 + ic) * 3 + dk];
        ch[oc] = a;
    }
    for (int oc = 0; oc < 21; ++oc) {            // k=5, pad 2 -> window 1..5
        float a = c1b[oc];
        for (int ic = 0; ic < 3; ++ic)
            for (int dk = 0; dk < 5; ++dk) a += win[ic][dk + 1] * c1w[(oc * 3 + ic) * 5 + dk];
        ch[22 + oc] = a;
    }
    for (int oc = 0; oc < 21; ++oc) {            // k=7, pad 3 -> window 0..6
        float a = c2b[oc];
        for (int ic = 0; ic < 3; ++ic)
            for (int dk = 0; dk < 7; ++dk) a += win[ic][dk] * c2w[(oc * 3 + ic) * 7 + dk];
        ch[43 + oc] = a;
    }
    float m = 0.f;
    for (int i = 0; i < 64; ++i) m += ch[i];
    m *= (1.0f / 64.0f);
    float v = 0.f;
    for (int i = 0; i < 64; ++i) { float d = ch[i] - m; v += d * d; }
    v *= (1.0f / 64.0f);
    float r = rsqrtf(v + 1e-5f);
    for (int i = 0; i < 64; ++i) {
        float y = gelu_f((ch[i] - m) * r * lng[i] + lnb[i]);
        beF[(size_t)t * 64 + i] = y;
        beB[(size_t)t * 64 + i] = (bf16_t)y;
    }
}

// ---------------- WMMA GEMM:  C[M,N] = A[M,K] @ W[N,K]^T (+bias) ----------------
// block = 8 waves; each wave computes a 32x64 tile (2x4 accumulators).
// K-loop unrolled x2 with ping-pong fragment buffers: no register rotation,
// loads of one chunk overlap WMMAs of the other. Requires K % 64 == 0.
__global__ __launch_bounds__(256)
void k_gemm(const bf16_t* __restrict__ A, int lda,
            const bf16_t* __restrict__ W, int ldw,
            const float* __restrict__ bias,
            float* __restrict__ Cf, bf16_t* __restrict__ Cb, int ldc,
            int M, int K, int accum) {
    int wave = threadIdx.x >> 5;
    int lane = threadIdx.x & 31;
    int hi = lane >> 4;
    int lr = lane & 15;
    int m0 = (blockIdx.x * 8 + wave) * 32;
    int n0 = blockIdx.y * 64;
    if (m0 >= M) return;
    v8f acc[2][4];
#pragma unroll
    for (int i = 0; i < 2; ++i)
#pragma unroll
        for (int j = 0; j < 4; ++j) acc[i][j] = v8f_zero();
    const bf16_t* Ar0 = A + (size_t)(m0 + lr) * lda;
    const bf16_t* Ar1 = A + (size_t)(m0 + 16 + lr) * lda;
    const bf16_t* Wb  = W + (size_t)(n0 + lr) * ldw + 16 * hi;
    size_t wrow = (size_t)16 * ldw;

    // buffer set A holds chunk k0; buffer set B holds chunk k0+32
    v16bf a0A = ld_afrag(Ar0, 0, hi);
    v16bf a1A = ld_afrag(Ar1, 0, hi);
    v16bf bA[4];
#pragma unroll
    for (int j = 0; j < 4; ++j) bA[j] = ld_bfrag(Wb + j * wrow);
    v16bf a0B, a1B, bBt[4];

    int k0 = 0;
    for (; k0 + 64 < K; k0 += 64) {
        a0B = ld_afrag(Ar0, k0 + 32, hi);
        a1B = ld_afrag(Ar1, k0 + 32, hi);
#pragma unroll
        for (int j = 0; j < 4; ++j) bBt[j] = ld_bfrag(Wb + j * wrow + k0 + 32);
#pragma unroll
        for (int j = 0; j < 4; ++j) {
            acc[0][j] = __builtin_amdgcn_wmma_f32_16x16x32_bf16(
                false, a0A, false, bA[j], (short)0, acc[0][j], false, false);
            acc[1][j] = __builtin_amdgcn_wmma_f32_16x16x32_bf16(
                false, a1A, false, bA[j], (short)0, acc[1][j], false, false);
        }
        a0A = ld_afrag(Ar0, k0 + 64, hi);
        a1A = ld_afrag(Ar1, k0 + 64, hi);
#pragma unroll
        for (int j = 0; j < 4; ++j) bA[j] = ld_bfrag(Wb + j * wrow + k0 + 64);
#pragma unroll
        for (int j = 0; j < 4; ++j) {
            acc[0][j] = __builtin_amdgcn_wmma_f32_16x16x32_bf16(
                false, a0B, false, bBt[j], (short)0, acc[0][j], false, false);
            acc[1][j] = __builtin_amdgcn_wmma_f32_16x16x32_bf16(
                false, a1B, false, bBt[j], (short)0, acc[1][j], false, false);
        }
    }
    // tail: chunks k0 (set A) and k0+32 (set B)
    a0B = ld_afrag(Ar0, k0 + 32, hi);
    a1B = ld_afrag(Ar1, k0 + 32, hi);
#pragma unroll
    for (int j = 0; j < 4; ++j) bBt[j] = ld_bfrag(Wb + j * wrow + k0 + 32);
#pragma unroll
    for (int j = 0; j < 4; ++j) {
        acc[0][j] = __builtin_amdgcn_wmma_f32_16x16x32_bf16(
            false, a0A, false, bA[j], (short)0, acc[0][j], false, false);
        acc[1][j] = __builtin_amdgcn_wmma_f32_16x16x32_bf16(
            false, a1A, false, bA[j], (short)0, acc[1][j], false, false);
    }
#pragma unroll
    for (int j = 0; j < 4; ++j) {
        acc[0][j] = __builtin_amdgcn_wmma_f32_16x16x32_bf16(
            false, a0B, false, bBt[j], (short)0, acc[0][j], false, false);
        acc[1][j] = __builtin_amdgcn_wmma_f32_16x16x32_bf16(
            false, a1B, false, bBt[j], (short)0, acc[1][j], false, false);
    }
#pragma unroll
    for (int j = 0; j < 4; ++j) {
        int n = n0 + j * 16 + lr;
        float bv = bias ? bias[n] : 0.0f;
#pragma unroll
        for (int i = 0; i < 2; ++i)
#pragma unroll
            for (int r = 0; r < 8; ++r) {
                int m = m0 + i * 16 + r + 8 * hi;
                size_t idx = (size_t)m * ldc + n;
                float v = acc[i][j][r] + bv;
                if (Cb) {
                    Cb[idx] = (bf16_t)v;
                } else {
                    if (accum) v += Cf[idx];
                    Cf[idx] = v;
                }
            }
    }
}

// ---------------- per-layer bloom bias MLP (64->8->4) ----------------
__global__ void k_bloombias(const float* __restrict__ beF,
    const float* __restrict__ w1, const float* __restrict__ b1,
    const float* __restrict__ w2, const float* __restrict__ b2,
    float* __restrict__ bb) {
    int t = blockIdx.x * blockDim.x + threadIdx.x;
    if (t >= TT) return;
    const float* x = beF + (size_t)t * 64;
    float h[8];
    for (int i = 0; i < 8; ++i) {
        float a = b1[i];
        for (int j = 0; j < 64; ++j) a += x[j] * w1[i * 64 + j];
        h[i] = gelu_f(a);
    }
    for (int o = 0; o < 4; ++o) {
        float a = b2[o];
        for (int i = 0; i < 8; ++i) a += h[i] * w2[o * 8 + i];
        bb[(size_t)t * 4 + o] = a;
    }
}

// ---------------- flash attention ----------------
// grid (S/128, H, B); block = 256 (8 waves); wave owns 16 queries.
__global__ __launch_bounds__(256, 1)
void k_attention(const bf16_t* __restrict__ Q, const bf16_t* __restrict__ Kg_,
                 const bf16_t* __restrict__ Vg_, const float* __restrict__ bb,
                 bf16_t* __restrict__ ctx) {
    __shared__ bf16_t ldsK[32 * HD];    // [key][dim] row-major
    __shared__ bf16_t ldsVt[HD * 32];   // [dim][key] transposed
    __shared__ float  ldsP[8][16 * 32]; // per-wave P staging
    int tid = threadIdx.x;
    int wave = tid >> 5, lane = tid & 31;
    int hi = lane >> 4, lr = lane & 15;
    int b = blockIdx.z, h = blockIdx.y;
    int q0 = blockIdx.x * 128 + wave * 16;
    const float rscale = 0.07216878364870323f; // 1/sqrt(192)

    v16bf qa[6];
    {
        const bf16_t* Qr = Q + ((size_t)(b * SS) + q0 + lr) * DM + h * HD;
#pragma unroll
        for (int c = 0; c < 6; ++c) qa[c] = ld_afrag(Qr, c * 32, hi);
    }
    v8f o[12];
#pragma unroll
    for (int j = 0; j < 12; ++j) o[j] = v8f_zero();
    float mrow[8], lrow[8];
#pragma unroll
    for (int r = 0; r < 8; ++r) { mrow[r] = -3.0e38f; lrow[r] = 0.0f; }

    for (int kt = 0; kt < SS; kt += 32) {
        const bf16_t* Kp = Kg_ + ((size_t)(b * SS) + kt) * DM + h * HD;
        const bf16_t* Vp = Vg_ + ((size_t)(b * SS) + kt) * DM + h * HD;
        __syncthreads();
        for (int idx = tid; idx < 32 * HD; idx += 256) {
            int key = idx / HD, d = idx % HD;
            ldsK[idx] = Kp[(size_t)key * DM + d];
            ldsVt[d * 32 + key] = Vp[(size_t)key * DM + d];
        }
        __syncthreads();
        if (kt + 32 < SS) {
            // 32 rows x 192 dims bf16 = 12 KB = 192 cachelines each; one line per thread
            if (tid < 192) {
                int key = tid >> 3, dd = (tid & 7) * 24;  // contiguous coverage
                __builtin_prefetch(Kp + (size_t)32 * DM + (size_t)key * DM + dd, 0, 0);
                __builtin_prefetch(Vp + (size_t)32 * DM + (size_t)key * DM + dd, 0, 0);
            }
        }
        // scores: two 16-key tiles, K-dim = 192; B-frags loaded 3-at-a-time
        v8f st[2];
#pragma unroll
        for (int t2 = 0; t2 < 2; ++t2) {
            v8f sa = v8f_zero();
            const bf16_t* Krow = &ldsK[(t2 * 16 + lr) * HD + 16 * hi];
#pragma unroll
            for (int cg = 0; cg < 2; ++cg) {
                v16bf bfg[3];
#pragma unroll
                for (int c = 0; c < 3; ++c) bfg[c] = ld_bfrag(Krow + (cg * 3 + c) * 32);
#pragma unroll
                for (int c = 0; c < 3; ++c)
                    sa = __builtin_amdgcn_wmma_f32_16x16x32_bf16(
                        false, qa[cg * 3 + c], false, bfg[c], (short)0, sa, false, false);
            }
            float biasv = bb[((size_t)(b * SS) + kt + t2 * 16 + lr) * 4 + h];
#pragma unroll
            for (int r = 0; r < 8; ++r) sa[r] = sa[r] * rscale + biasv;
            st[t2] = sa;
        }
        // online softmax (lane holds one column; rows r+8*hi)
        float p0[8], p1[8], scl[8];
#pragma unroll
        for (int r = 0; r < 8; ++r) {
            float v = fmaxf(st[0][r], st[1][r]);
            v = fmaxf(v, __shfl_xor(v, 1, 32));
            v = fmaxf(v, __shfl_xor(v, 2, 32));
            v = fmaxf(v, __shfl_xor(v, 4, 32));
            v = fmaxf(v, __shfl_xor(v, 8, 32));
            float mnew = fmaxf(mrow[r], v);
            scl[r] = expf(mrow[r] - mnew);
            mrow[r] = mnew;
            p0[r] = expf(st[0][r] - mnew);
            p1[r] = expf(st[1][r] - mnew);
            float s = p0[r] + p1[r];
            s += __shfl_xor(s, 1, 32);
            s += __shfl_xor(s, 2, 32);
            s += __shfl_xor(s, 4, 32);
            s += __shfl_xor(s, 8, 32);
            lrow[r] = lrow[r] * scl[r] + s;
        }
#pragma unroll
        for (int j = 0; j < 12; ++j)
#pragma unroll
            for (int r = 0; r < 8; ++r) o[j][r] *= scl[r];
        // stage P (C layout) -> LDS row-major [16][32]
#pragma unroll
        for (int r = 0; r < 8; ++r) {
            ldsP[wave][(r + 8 * hi) * 32 + lr]      = p0[r];
            ldsP[wave][(r + 8 * hi) * 32 + 16 + lr] = p1[r];
        }
        // reload as A fragment (16x32)
        v16bf pa;
#pragma unroll
        for (int e = 0; e < 16; ++e) {
            int k = e + 16 * hi;
            pa[e] = (bf16_t)ldsP[wave][lr * 32 + k];
        }
        // O += P @ V ; B-frags from transposed V, loaded 4-at-a-time
        const bf16_t* Vrow = &ldsVt[lr * 32 + 16 * hi];
#pragma unroll
        for (int jg = 0; jg < 3; ++jg) {
            v16bf bfg[4];
#pragma unroll
            for (int j = 0; j < 4; ++j) bfg[j] = ld_bfrag(Vrow + (jg * 4 + j) * (16 * 32));
#pragma unroll
            for (int j = 0; j < 4; ++j)
                o[jg * 4 + j] = __builtin_amdgcn_wmma_f32_16x16x32_bf16(
                    false, pa, false, bfg[j], (short)0, o[jg * 4 + j], false, false);
        }
    }
    float invl[8];
#pragma unroll
    for (int r = 0; r < 8; ++r) invl[r] = 1.0f / lrow[r];
#pragma unroll
    for (int j = 0; j < 12; ++j)
#pragma unroll
        for (int r = 0; r < 8; ++r) {
            float val = o[j][r] * invl[r];
            ctx[((size_t)(b * SS) + q0 + r + 8 * hi) * DM + h * HD + j * 16 + lr] = (bf16_t)val;
        }
}

// ---------------- residual + LayerNorm ----------------
__global__ void k_resln(const float* __restrict__ x, const float* __restrict__ res,
                        const float* __restrict__ g, const float* __restrict__ bta,
                        float* __restrict__ hF, bf16_t* __restrict__ hB) {
    int t = blockIdx.x * blockDim.x + threadIdx.x;
    if (t >= TT) return;
    const float* xr = x + (size_t)t * DM;
    const float* rr = res + (size_t)t * DM;
    float m = 0.f, v = 0.f;
    for (int d = 0; d < DM; ++d) { float a = xr[d] + rr[d]; m += a; v += a * a; }
    m /= (float)DM;
    v = v / (float)DM - m * m;
    float rs = rsqrtf(v + 1e-5f);
    for (int d = 0; d < DM; ++d) {
        float a = xr[d] + rr[d];
        float y = (a - m) * rs * g[d] + bta[d];
        hF[(size_t)t * DM + d] = y;
        hB[(size_t)t * DM + d] = (bf16_t)y;
    }
}

// ---------------- pooling: gelu(hidden)·w2 ----------------
__global__ void k_imp(const float* __restrict__ H, const float* __restrict__ w2,
                      const float* __restrict__ b2, float* __restrict__ imp) {
    int t = blockIdx.x * blockDim.x + threadIdx.x;
    if (t >= TT) return;
    const float* hr = H + (size_t)t * 256;
    float s = b2[0];
    for (int j = 0; j < 256; ++j) s += gelu_f(hr[j]) * w2[j];
    imp[t] = s;
}

__global__ void k_softmax_s(const float* __restrict__ imp, float* __restrict__ wts) {
    __shared__ float red[256];
    int b = blockIdx.x, tid = threadIdx.x;
    const float* x = imp + (size_t)b * SS;
    float mx = -3.0e38f;
    for (int s = tid; s < SS; s += 256) mx = fmaxf(mx, x[s]);
    red[tid] = mx; __syncthreads();
    for (int o = 128; o > 0; o >>= 1) { if (tid < o) red[tid] = fmaxf(red[tid], red[tid + o]); __syncthreads(); }
    mx = red[0]; __syncthreads();
    float sum = 0.f;
    for (int s = tid; s < SS; s += 256) sum += expf(x[s] - mx);
    red[tid] = sum; __syncthreads();
    for (int o = 128; o > 0; o >>= 1) { if (tid < o) red[tid] += red[tid + o]; __syncthreads(); }
    float inv = 1.0f / red[0];
    for (int s = tid; s < SS; s += 256) wts[(size_t)b * SS + s] = expf(x[s] - mx) * inv;
}

__global__ void k_pool(const float* __restrict__ wts, const float* __restrict__ h,
                       float* __restrict__ pooled) {
    int b = blockIdx.y;
    int d = blockIdx.x * 256 + threadIdx.x;
    float acc = 0.f;
    for (int s = 0; s < SS; ++s) acc += wts[(size_t)b * SS + s] * h[((size_t)b * SS + s) * DM + d];
    pooled[(size_t)b * DM + d] = acc;
}

// ---------------- fusion + head (per batch) ----------------
__global__ __launch_bounds__(256)
void k_fuse_head(const float* __restrict__ pooled, const float* __restrict__ summ,
    const float* __restrict__ gw1, const float* __restrict__ gb1,
    const float* __restrict__ gw2, const float* __restrict__ gb2,
    const float* __restrict__ pw1, const float* __restrict__ pb1,
    const float* __restrict__ pw2, const float* __restrict__ pb2,
    const float* __restrict__ flng, const float* __restrict__ flnb,
    const float* __restrict__ hw1, const float* __restrict__ hb1,
    const float* __restrict__ hw2, const float* __restrict__ hb2,
    float* __restrict__ out) {
    __shared__ float sP[DM], sG[DM], sF[DM], sH[256], red[256], sSum[18];
    int b = blockIdx.x, tid = threadIdx.x;
    for (int d = tid; d < DM; d += 256) sP[d] = pooled[(size_t)b * DM + d];
    if (tid < 18) sSum[tid] = summ[b * 18 + tid];
    __syncthreads();
    { // gate hidden (256)
        float a = gb1[tid];
        const float* wr = gw1 + (size_t)tid * (DM + 18);
        for (int i = 0; i < DM; ++i) a += sP[i] * wr[i];
        for (int i = 0; i < 18; ++i) a += sSum[i] * wr[DM + i];
        sH[tid] = gelu_f(a);
    }
    __syncthreads();
    for (int d = tid; d < DM; d += 256) { // gate output (768) + sigmoid
        float a = gb2[d];
        const float* wr = gw2 + (size_t)d * 256;
        for (int j = 0; j < 256; ++j) a += sH[j] * wr[j];
        sG[d] = 1.0f / (1.0f + expf(-a));
    }
    __syncthreads();
    if (tid < 128) { // bproj hidden (128)
        float a = pb1[tid];
        const float* wr = pw1 + (size_t)tid * 18;
        for (int i = 0; i < 18; ++i) a += sSum[i] * wr[i];
        sH[tid] = gelu_f(a);
    }
    __syncthreads();
    for (int d = tid; d < DM; d += 256) { // bproj out + gated mix
        float a = pb2[d];
        const float* wr = pw2 + (size_t)d * 128;
        for (int j = 0; j < 128; ++j) a += sH[j] * wr[j];
        sF[d] = sG[d] * sP[d] + (1.0f - sG[d]) * a;
    }
    __syncthreads();
    // LayerNorm over sF
    float pm = 0.f;
    for (int d = tid; d < DM; d += 256) pm += sF[d];
    red[tid] = pm; __syncthreads();
    for (int o = 128; o > 0; o >>= 1) { if (tid < o) red[tid] += red[tid + o]; __syncthreads(); }
    float mean = red[0] / (float)DM; __syncthreads();
    float pv = 0.f;
    for (int d = tid; d < DM; d += 256) { float dd = sF[d] - mean; pv += dd * dd; }
    red[tid] = pv; __syncthreads();
    for (int o = 128; o > 0; o >>= 1) { if (tid < o) red[tid] += red[tid + o]; __syncthreads(); }
    float rs = rsqrtf(red[0] / (float)DM + 1e-5f);
    __syncthreads();
    for (int d = tid; d < DM; d += 256) sF[d] = (sF[d] - mean) * rs * flng[d] + flnb[d];
    __syncthreads();
    { // head: 768 -> 256 gelu -> 1
        float a = hb1[tid];
        const float* wr = hw1 + (size_t)tid * DM;
        for (int d = 0; d < DM; ++d) a += sF[d] * wr[d];
        red[tid] = gelu_f(a) * hw2[tid];
    }
    __syncthreads();
    for (int o = 128; o > 0; o >>= 1) { if (tid < o) red[tid] += red[tid + o]; __syncthreads(); }
    if (tid == 0) out[b] = red[0] + hb2[0];
}

// ---------------- host launch ----------------
extern "C" void kernel_launch(void* const* d_in, const int* in_sizes, int n_in,
                              void* d_out, int out_size, void* d_ws, size_t ws_size,
                              hipStream_t stream) {
    (void)in_sizes; (void)n_in; (void)out_size; (void)ws_size;
    const float* dnah = (const float*)d_in[0];
    const float* sig  = (const float*)d_in[1];
    const float* summ = (const float*)d_in[2];
    auto P = [&](int i) -> const float* { return (const float*)d_in[3 + i]; };
    // param layout (relative): 0 scale_w, 1-6 conv w/b x3, 7 enc_ln_g, 8 enc_ln_b,
    // layer l base = 9+14*l: +0 Wq +1 bq +2 Wk +3 bk +4 Wv +5 bv +6 bp1w +7 bp1b
    //                        +8 bp2w +9 bp2b +10 Wo +11 bo +12 ln_g +13 ln_b
    // 37 pool_w1, 38 pool_b1, 39 pool_w2, 40 pool_b2, 41-44 gate, 45-48 bproj,
    // 49-50 fuse_ln, 51-54 head.

    char* wsb = (char*)d_ws;
    size_t off = 0;
    auto alloc = [&](size_t bytes) -> void* {
        void* p = wsb + off;
        off = (off + bytes + 255) & ~(size_t)255;
        return p;
    };
    float*  hF    = (float*) alloc((size_t)TT * DM * 4);
    bf16_t* hB    = (bf16_t*)alloc((size_t)TT * DM * 2);
    float*  beF   = (float*) alloc((size_t)TT * DB * 4);
    bf16_t* beB   = (bf16_t*)alloc((size_t)TT * DB * 2);
    bf16_t* qB    = (bf16_t*)alloc((size_t)TT * DM * 2);
    bf16_t* kB    = (bf16_t*)alloc((size_t)TT * DM * 2);
    bf16_t* vB    = (bf16_t*)alloc((size_t)TT * DM * 2);
    bf16_t* ctxB  = (bf16_t*)alloc((size_t)TT * DM * 2);
    float*  attnF = (float*) alloc((size_t)TT * DM * 4);
    float*  bbF   = (float*) alloc((size_t)TT * 4 * 4);
    float*  impF  = (float*) alloc((size_t)TT * 4);
    float*  wtsF  = (float*) alloc((size_t)TT * 4);
    float*  poolF = (float*) alloc((size_t)BB * DM * 4);
    bf16_t* wqB[2], *wkB[2], *wvB[2], *woB[2];
    for (int l = 0; l < 2; ++l) {
        wqB[l] = (bf16_t*)alloc((size_t)DM * DM * 2);
        wkB[l] = (bf16_t*)alloc((size_t)DM * DB * 2);
        wvB[l] = (bf16_t*)alloc((size_t)DM * DM * 2);
        woB[l] = (bf16_t*)alloc((size_t)DM * DM * 2);
    }
    bf16_t* pw1B = (bf16_t*)alloc((size_t)256 * (DM + DB) * 2);
    float* impH = attnF; // reuse (TT*256*4 <= TT*DM*4), free after last res-LN

    auto cvt = [&](const float* s, bf16_t* d, size_t n) {
        k_cvt_bf16<<<(unsigned)((n + 255) / 256), 256, 0, stream>>>(s, d, (int)n);
    };
    for (int l = 0; l < 2; ++l) {
        int base = 9 + 14 * l;
        cvt(P(base + 0),  wqB[l], (size_t)DM * DM);
        cvt(P(base + 2),  wkB[l], (size_t)DM * DB);
        cvt(P(base + 4),  wvB[l], (size_t)DM * DM);
        cvt(P(base + 10), woB[l], (size_t)DM * DM);
    }
    cvt(P(37), pw1B, (size_t)256 * (DM + DB));
    k_copy_cvt<<<(TT * DM) / 256, 256, 0, stream>>>(dnah, hF, hB, TT * DM);
    k_encoder<<<TT / 256, 256, 0, stream>>>(sig, P(0), P(1), P(2), P(3), P(4), P(5), P(6),
                                            P(7), P(8), beF, beB);
    dim3 gG(TT / 256, DM / 64);
    for (int l = 0; l < 2; ++l) {
        int base = 9 + 14 * l;
        k_gemm<<<gG, 256, 0, stream>>>(hB,  DM, wqB[l], DM, P(base + 1), nullptr, qB, DM, TT, DM, 0);
        k_gemm<<<gG, 256, 0, stream>>>(beB, DB, wkB[l], DB, P(base + 3), nullptr, kB, DM, TT, DB, 0);
        k_gemm<<<gG, 256, 0, stream>>>(hB,  DM, wvB[l], DM, P(base + 5), nullptr, vB, DM, TT, DM, 0);
        k_bloombias<<<TT / 256, 256, 0, stream>>>(beF, P(base + 6), P(base + 7),
                                                  P(base + 8), P(base + 9), bbF);
        dim3 gA(SS / 128, NH, BB);
        k_attention<<<gA, 256, 0, stream>>>(qB, kB, vB, bbF, ctxB);
        k_gemm<<<gG, 256, 0, stream>>>(ctxB, DM, woB[l], DM, P(base + 11), attnF, nullptr, DM, TT, DM, 0);
        k_resln<<<TT / 256, 256, 0, stream>>>(attnF, hF, P(base + 12), P(base + 13), hF, hB);
    }
    // pooling: impH = h @ pool_w1[:, :768]^T ; += be @ pool_w1[:, 768:]^T + pool_b1
    dim3 gP(TT / 256, 256 / 64);
    k_gemm<<<gP, 256, 0, stream>>>(hB,  DM, pw1B,      DM + DB, nullptr, impH, nullptr, 256, TT, DM, 0);
    k_gemm<<<gP, 256, 0, stream>>>(beB, DB, pw1B + DM, DM + DB, P(38),   impH, nullptr, 256, TT, DB, 1);
    k_imp<<<TT / 256, 256, 0, stream>>>(impH, P(39), P(40), impF);
    k_softmax_s<<<BB, 256, 0, stream>>>(impF, wtsF);
    dim3 gPool(DM / 256, BB);
    k_pool<<<gPool, 256, 0, stream>>>(wtsF, hF, poolF);
    k_fuse_head<<<BB, 256, 0, stream>>>(poolF, summ, P(41), P(42), P(43), P(44),
        P(45), P(46), P(47), P(48), P(49), P(50), P(51), P(52), P(53), P(54),
        (float*)d_out);
}
